// VanillaFlashDiff1_30674656428184
// MI455X (gfx1250) — compile-verified, matrix operand order checked by reading, not run on hardware
//
#include <hip/hip_runtime.h>

// Problem constants (match reference)
#define B_  2
#define S_  2048
#define E_  2048
#define H_  16
#define D_  128
#define NEG_INF (-3.0e38f)

typedef __attribute__((ext_vector_type(16))) _Float16 v16h;
typedef __attribute__((ext_vector_type(8)))  float    v8f;
typedef __attribute__((ext_vector_type(4)))  float    f32x4;
typedef __attribute__((ext_vector_type(8)))  _Float16 h8;
typedef __attribute__((ext_vector_type(4)))  int      i32x4;

#if __has_builtin(__builtin_amdgcn_global_load_async_to_lds_b128)
#define ASYNC_LDS 1
#else
#define ASYNC_LDS 0
#endif

__device__ __forceinline__ v8f zero8() {
  v8f z;
#pragma unroll
  for (int i = 0; i < 8; ++i) z[i] = 0.0f;
  return z;
}

// Convert 8 f32 (two f32x4 already in registers) to f16 and store 16B to LDS.
__device__ __forceinline__ void pack_store8(_Float16* dst, f32x4 a, f32x4 b) {
  h8 hv;
  hv[0] = (_Float16)a[0]; hv[1] = (_Float16)a[1];
  hv[2] = (_Float16)a[2]; hv[3] = (_Float16)a[3];
  hv[4] = (_Float16)b[0]; hv[5] = (_Float16)b[1];
  hv[6] = (_Float16)b[2]; hv[7] = (_Float16)b[3];
  *reinterpret_cast<h8*>(dst) = hv;
}

// f16 16B staging global->LDS. Prefer CDNA5 async-to-LDS (ASYNCcnt-tracked,
// VGPR-bypassing DMA); fall back to a plain vector copy.
// Builtin signature: (global i32x4*, lds i32x4*, imm offset, imm cpol).
__device__ __forceinline__ void stage16(_Float16* lds, const _Float16* gsrc) {
#if ASYNC_LDS
  __builtin_amdgcn_global_load_async_to_lds_b128(
      (__attribute__((address_space(1))) i32x4*)gsrc,
      (__attribute__((address_space(3))) i32x4*)lds,
      0, 0);
#else
  *reinterpret_cast<h8*>(lds) = *reinterpret_cast<const h8*>(gsrc);
#endif
}

__device__ __forceinline__ void wait_async() {
#if ASYNC_LDS
#if __has_builtin(__builtin_amdgcn_s_wait_asynccnt)
  __builtin_amdgcn_s_wait_asynccnt(0);
#else
  asm volatile("s_wait_asynccnt 0x0" ::: "memory");
#endif
#endif
}

// Generic 16-wide f16 fragment loader following CDNA5 ISA 7.12.2 (wave32):
//   element index = (lane&15)*sLane + kpat*sK
//   lanes 0-15 hold K = 0..7 / 16..23 ; lanes 16-31 hold K = 8..15 / 24..31
// A-matrix use: sLane = row stride, sK = K stride.
// B-matrix use: sLane = col (N) stride, sK = K stride.
__device__ __forceinline__ v16h frag16(const _Float16* base, int sLane, int sK, int lane) {
  const int l  = lane & 15;
  const int kb = (lane & 16) ? 8 : 0;
  const _Float16* p = base + (size_t)l * sLane;
  v16h f;
#pragma unroll
  for (int i = 0; i < 4; ++i) {
    f[2*i]     = p[(kb + 2*i)          * sK];
    f[2*i + 1] = p[(kb + 2*i + 1)      * sK];
    f[8 + 2*i]     = p[(kb + 16 + 2*i)     * sK];
    f[8 + 2*i + 1] = p[(kb + 16 + 2*i + 1) * sK];
  }
  return f;
}

__device__ __forceinline__ v8f wmma_f16(v16h a, v16h b, v8f c) {
  // D = A(16x32 f16) x B(32x16 f16) + C(16x16 f32)
  return __builtin_amdgcn_wmma_f32_16x16x32_f16(false, a, false, b, (short)0, c, false, false);
}

// ---------------------------------------------------------------------------
// Kernel 1: fused QKV projection + interleaved RoPE epilogue.
// C[m,n] = sum_k x[m,k] * W[n,k]  (x @ W^T), output f16, layout [B][H][S][D].
// gridDim.z selects Q / K / V.
// ---------------------------------------------------------------------------
__global__ __launch_bounds__(256) void qkv_rope_kernel(
    const float* __restrict__ x,  const float* __restrict__ Wq,
    const float* __restrict__ Wk, const float* __restrict__ Wv,
    _Float16* __restrict__ qh, _Float16* __restrict__ kh, _Float16* __restrict__ vh)
{
  __shared__ _Float16 As[128][40];   // x tile [m][k], +8 pad (80B rows, 16B aligned)
  __shared__ _Float16 Bs[128][40];   // W tile [n][k]

  const int n0 = blockIdx.x * 128;
  const int m0 = blockIdx.y * 128;
  const int which = blockIdx.z;                 // 0=Q 1=K 2=V (uniform per block)
  const float* W = (which == 0) ? Wq : (which == 1) ? Wk : Wv;
  _Float16* dst  = (which == 0) ? qh : (which == 1) ? kh : vh;

  const int tid  = threadIdx.x;
  const int lane = tid & 31;
  const int wave = tid >> 5;      // 8 waves
  const int wm   = wave >> 1;     // 0..3 -> 32-row slab
  const int wn   = wave & 1;      // 0..1 -> 64-col slab

  v8f acc[2][4];
#pragma unroll
  for (int mi = 0; mi < 2; ++mi)
#pragma unroll
    for (int ni = 0; ni < 4; ++ni) acc[mi][ni] = zero8();

  for (int k0 = 0; k0 < E_; k0 += 32) {
    // Stage tiles f32 -> f16. Issue ALL global loads first (one clause, loads
    // in flight together), then convert + ds_store — avoids per-chunk
    // s_wait_loadcnt 0 drains.
    f32x4 xa[2][2], wb[2][2];
#pragma unroll
    for (int i = 0; i < 2; ++i) {
      int chunk = i * 256 + tid;          // 512 chunks of 8 elements
      int r  = chunk >> 2;                // 4 chunks per row
      int cc = (chunk & 3) * 8;
      const f32x4* xs = reinterpret_cast<const f32x4*>(x + (size_t)(m0 + r) * E_ + k0 + cc);
      const f32x4* ws = reinterpret_cast<const f32x4*>(W + (size_t)(n0 + r) * E_ + k0 + cc);
      xa[i][0] = xs[0]; xa[i][1] = xs[1];
      wb[i][0] = ws[0]; wb[i][1] = ws[1];
    }
#pragma unroll
    for (int i = 0; i < 2; ++i) {
      int chunk = i * 256 + tid;
      int r  = chunk >> 2;
      int cc = (chunk & 3) * 8;
      pack_store8(&As[r][cc], xa[i][0], xa[i][1]);
      pack_store8(&Bs[r][cc], wb[i][0], wb[i][1]);
    }
    if (k0 + 32 < E_) {
      // hint the next K-slab into cache (global_prefetch_b8)
      __builtin_prefetch(x + (size_t)(m0 + (tid >> 1)) * E_ + k0 + 32, 0, 3);
      __builtin_prefetch(W + (size_t)(n0 + (tid >> 1)) * E_ + k0 + 32, 0, 3);
    }
    __syncthreads();

    v16h a[2];
#pragma unroll
    for (int mi = 0; mi < 2; ++mi)
      a[mi] = frag16(&As[wm * 32 + mi * 16][0], 40, 1, lane);
#pragma unroll
    for (int ni = 0; ni < 4; ++ni) {
      v16h b = frag16(&Bs[wn * 64 + ni * 16][0], 40, 1, lane);
#pragma unroll
      for (int mi = 0; mi < 2; ++mi)
        acc[mi][ni] = wmma_f16(a[mi], b, acc[mi][ni]);
    }
    __syncthreads();
  }

  // Epilogue: RoPE (interleaved) for Q/K, then store f16 head-major.
  // C-layout: lanes 0-15 row = r, lanes 16-31 row = r+8; col = lane&15.
  // Pair (even,odd) columns sit in lanes (2p,2p+1) -> shfl_xor(1) exchange.
#pragma unroll
  for (int mi = 0; mi < 2; ++mi) {
#pragma unroll
    for (int ni = 0; ni < 4; ++ni) {
#pragma unroll
      for (int r = 0; r < 8; ++r) {
        int rowl = (lane & 16) ? (r + 8) : r;
        int gm = m0 + wm * 32 + mi * 16 + rowl;
        int gn = n0 + wn * 64 + ni * 16 + (lane & 15);
        int bb = gm >> 11;            // / S_
        int ss = gm & (S_ - 1);
        int hh = gn >> 7;             // / D_
        int dd = gn & (D_ - 1);
        float val = acc[mi][ni][r];
        float out;
        if (which < 2) {
          float other = __shfl_xor(val, 1, 32);
          int j = dd >> 1;
          // inv_freq = 10000^(-2j/128) = exp(-ln(10000)/64 * j)
          float ang = (float)ss * __expf(-0.14391157f * (float)j);
          float c, s;
          __sincosf(ang, &s, &c);
          out = (dd & 1) ? (other * s + val * c)    // o2 = x1*s + x2*c
                         : (val * c - other * s);   // o1 = x1*c - x2*s
        } else {
          out = val;
        }
        dst[(((size_t)bb * H_ + hh) * S_ + ss) * D_ + dd] = (_Float16)out;
      }
    }
  }
}

// ---------------------------------------------------------------------------
// Kernel 2: causal flash attention per (b,h). BQ=64 per block, 4 waves.
// Streams K/V tiles of 64 through LDS (async-to-LDS DMA), online softmax,
// WMMA for QK^T and PV.
// Output f16, layout [B][S][H*D] (ready for the Wo projection).
// ---------------------------------------------------------------------------
__global__ __launch_bounds__(128) void attn_kernel(
    const _Float16* __restrict__ qh, const _Float16* __restrict__ kh,
    const _Float16* __restrict__ vh, _Float16* __restrict__ attnh)
{
  __shared__ _Float16 Kt[64][136];      // [kv][d], 272B rows (16B aligned)
  __shared__ _Float16 Vt[64][136];
  __shared__ _Float16 Pb[4][16][72];    // per-wave P staging (C-layout -> A-layout)

  const int q0  = blockIdx.x * 64;
  const int bh  = blockIdx.y;
  const int b   = bh >> 4;              // / H_
  const int h   = bh & 15;
  const int tid = threadIdx.x;
  const int lane = tid & 31;
  const int wave = tid >> 5;            // 4 waves, each owns 16 query rows

  const size_t headOff = ((size_t)b * H_ + h) * S_;
  const _Float16* qp = qh + (headOff + q0 + wave * 16) * D_;

  // Q fragments (16 x 128) resident in registers: 4 K-chunks of 32
  v16h Qf[4];
#pragma unroll
  for (int dc = 0; dc < 4; ++dc)
    Qf[dc] = frag16(qp + dc * 32, D_, 1, lane);

  float m_i[8], l_i[8], alpha[8];
  v8f acc[8];                           // O accumulator: 8 d-tiles of 16
#pragma unroll
  for (int r = 0; r < 8; ++r) { m_i[r] = NEG_INF; l_i[r] = 0.0f; }
#pragma unroll
  for (int dt = 0; dt < 8; ++dt) acc[dt] = zero8();

  const float scale = 0.08838834764831845f;   // 1/sqrt(128)

  for (int kv0 = 0; kv0 <= q0; kv0 += 64) {   // causal: stop at diagonal tile
    // stage K/V tiles (64x128 f16): async DMA to LDS
#pragma unroll
    for (int i = 0; i < 8; ++i) {
      int chunk = i * 128 + tid;              // 1024 chunks of 8 halfs
      int r  = chunk >> 4;
      int cc = (chunk & 15) * 8;
      const size_t src = (headOff + kv0 + r) * D_ + cc;
      stage16(&Kt[r][cc], kh + src);
      stage16(&Vt[r][cc], vh + src);
    }
    wait_async();
    __syncthreads();

    // S = Q K^T : 16 x 64 per wave. B element (k=d, n=key) from Kt[key][d].
    v8f sacc[4];
#pragma unroll
    for (int nt = 0; nt < 4; ++nt) {
      v8f s = zero8();
#pragma unroll
      for (int dc = 0; dc < 4; ++dc) {
        v16h bfr = frag16(&Kt[nt * 16][dc * 32], 136, 1, lane);
        s = wmma_f16(Qf[dc], bfr, s);
      }
      sacc[nt] = s;
    }

    // Online softmax. Row r lives across 16 lanes of each half-wave in slot r.
#pragma unroll
    for (int r = 0; r < 8; ++r) {
      int rowl = (lane & 16) ? (r + 8) : r;
      int qs = q0 + wave * 16 + rowl;
      float sv[4];
      float mv = NEG_INF;
#pragma unroll
      for (int nt = 0; nt < 4; ++nt) {
        int col = kv0 + nt * 16 + (lane & 15);
        float t = sacc[nt][r] * scale;
        if (col > qs) t = NEG_INF;            // causal mask
        sv[nt] = t;
        mv = fmaxf(mv, t);
      }
#pragma unroll
      for (int off = 8; off >= 1; off >>= 1)
        mv = fmaxf(mv, __shfl_xor(mv, off, 16));
      float mnew = fmaxf(m_i[r], mv);
      float a = __expf(m_i[r] - mnew);
      float rs = 0.0f;
#pragma unroll
      for (int nt = 0; nt < 4; ++nt) {
        float p = __expf(sv[nt] - mnew);
        rs += p;
        Pb[wave][rowl][nt * 16 + (lane & 15)] = (_Float16)p;
      }
#pragma unroll
      for (int off = 8; off >= 1; off >>= 1)
        rs += __shfl_xor(rs, off, 16);
      l_i[r] = l_i[r] * a + rs;
      m_i[r] = mnew;
      alpha[r] = a;
    }

    // rescale running O
#pragma unroll
    for (int dt = 0; dt < 8; ++dt)
#pragma unroll
      for (int r = 0; r < 8; ++r)
        acc[dt][r] *= alpha[r];

    // O += P V. P re-read from our own LDS region (same-wave DS ordering).
    v16h pa[2];
#pragma unroll
    for (int kc = 0; kc < 2; ++kc)
      pa[kc] = frag16(&Pb[wave][0][kc * 32], 72, 1, lane);
#pragma unroll
    for (int dt = 0; dt < 8; ++dt) {
#pragma unroll
      for (int kc = 0; kc < 2; ++kc) {
        // B element (k=kv, n=d) from Vt[kv][d]: col stride 1, K stride 136
        v16h bfr = frag16(&Vt[kc * 32][dt * 16], 1, 136, lane);
        acc[dt] = wmma_f16(pa[kc], bfr, acc[dt]);
      }
    }
    __syncthreads();   // protect Kt/Vt before next tile's staging
  }

  // normalize + store [B][S][H*D] f16
#pragma unroll
  for (int dt = 0; dt < 8; ++dt) {
#pragma unroll
    for (int r = 0; r < 8; ++r) {
      int rowl = (lane & 16) ? (r + 8) : r;
      int qs = q0 + wave * 16 + rowl;
      float rl = 1.0f / l_i[r];
      int col = h * D_ + dt * 16 + (lane & 15);
      attnh[((size_t)b * S_ + qs) * E_ + col] = (_Float16)(acc[dt][r] * rl);
    }
  }
}

// ---------------------------------------------------------------------------
// Kernel 3: output projection out = attn @ Wo^T, f32 result.
// ---------------------------------------------------------------------------
__global__ __launch_bounds__(256) void outproj_kernel(
    const _Float16* __restrict__ attnh, const float* __restrict__ Wo,
    float* __restrict__ out)
{
  __shared__ _Float16 As[128][40];
  __shared__ _Float16 Bs[128][40];

  const int n0 = blockIdx.x * 128;
  const int m0 = blockIdx.y * 128;
  const int tid  = threadIdx.x;
  const int lane = tid & 31;
  const int wave = tid >> 5;
  const int wm = wave >> 1;
  const int wn = wave & 1;

  v8f acc[2][4];
#pragma unroll
  for (int mi = 0; mi < 2; ++mi)
#pragma unroll
    for (int ni = 0; ni < 4; ++ni) acc[mi][ni] = zero8();

  for (int k0 = 0; k0 < E_; k0 += 32) {
    // A tile already f16: async-to-LDS. B tile f32->f16: load-all-then-convert.
    f32x4 wb[2][2];
#pragma unroll
    for (int i = 0; i < 2; ++i) {
      int chunk = i * 256 + tid;              // 512 chunks of 8 elements
      int r  = chunk >> 2;
      int cc = (chunk & 3) * 8;
      stage16(&As[r][cc], attnh + (size_t)(m0 + r) * E_ + k0 + cc);
      const f32x4* ws = reinterpret_cast<const f32x4*>(Wo + (size_t)(n0 + r) * E_ + k0 + cc);
      wb[i][0] = ws[0]; wb[i][1] = ws[1];
    }
#pragma unroll
    for (int i = 0; i < 2; ++i) {
      int chunk = i * 256 + tid;
      int r  = chunk >> 2;
      int cc = (chunk & 3) * 8;
      pack_store8(&Bs[r][cc], wb[i][0], wb[i][1]);
    }
    if (k0 + 32 < E_)
      __builtin_prefetch(Wo + (size_t)(n0 + (tid >> 1)) * E_ + k0 + 32, 0, 3);
    wait_async();
    __syncthreads();

    v16h a[2];
#pragma unroll
    for (int mi = 0; mi < 2; ++mi)
      a[mi] = frag16(&As[wm * 32 + mi * 16][0], 40, 1, lane);
#pragma unroll
    for (int ni = 0; ni < 4; ++ni) {
      v16h b = frag16(&Bs[wn * 64 + ni * 16][0], 40, 1, lane);
#pragma unroll
      for (int mi = 0; mi < 2; ++mi)
        acc[mi][ni] = wmma_f16(a[mi], b, acc[mi][ni]);
    }
    __syncthreads();
  }

#pragma unroll
  for (int mi = 0; mi < 2; ++mi) {
#pragma unroll
    for (int ni = 0; ni < 4; ++ni) {
#pragma unroll
      for (int r = 0; r < 8; ++r) {
        int rowl = (lane & 16) ? (r + 8) : r;
        int gm = m0 + wm * 32 + mi * 16 + rowl;
        int gn = n0 + wn * 64 + ni * 16 + (lane & 15);
        out[(size_t)gm * E_ + gn] = acc[mi][ni][r];
      }
    }
  }
}

// ---------------------------------------------------------------------------
extern "C" void kernel_launch(void* const* d_in, const int* in_sizes, int n_in,
                              void* d_out, int out_size, void* d_ws, size_t ws_size,
                              hipStream_t stream) {
  (void)in_sizes; (void)n_in; (void)out_size; (void)ws_size;
  const float* x  = (const float*)d_in[0];
  const float* Wq = (const float*)d_in[1];
  const float* Wk = (const float*)d_in[2];
  const float* Wv = (const float*)d_in[3];
  const float* Wo = (const float*)d_in[4];
  float* out = (float*)d_out;

  // Workspace: 4 x 16 MiB f16 buffers (q,k,v head-major; attn row-major)
  char* ws = (char*)d_ws;
  const size_t perBuf = (size_t)B_ * H_ * S_ * D_ * sizeof(_Float16);
  _Float16* qh    = (_Float16*)(ws);
  _Float16* kh    = (_Float16*)(ws + perBuf);
  _Float16* vh    = (_Float16*)(ws + 2 * perBuf);
  _Float16* attnh = (_Float16*)(ws + 3 * perBuf);

  qkv_rope_kernel<<<dim3(E_ / 128, (B_ * S_) / 128, 3), dim3(256), 0, stream>>>(
      x, Wq, Wk, Wv, qh, kh, vh);
  attn_kernel<<<dim3(S_ / 64, B_ * H_), dim3(128), 0, stream>>>(qh, kh, vh, attnh);
  outproj_kernel<<<dim3(E_ / 128, (B_ * S_) / 128), dim3(256), 0, stream>>>(attnh, Wo, out);
}
